// AttentionBlock_17059610100084
// MI455X (gfx1250) — compile-verified, matrix operand order checked by reading, not run on hardware
//
#include <hip/hip_runtime.h>
#include <hip/hip_bf16.h>
#include <stdint.h>

// Problem constants (from reference): x [B=16, C=512, H=32, W=32], groups=8
#define B_   16
#define C_   512
#define HW_  1024          // N tokens = H*W
#define G_   8
#define CG_  64            // C / G
#define EPS_ 1e-5f

typedef __attribute__((ext_vector_type(16))) __bf16 v16bf;
typedef __attribute__((ext_vector_type(8)))  float  v8f;

// ---------------------------------------------------------------- helpers ---

__device__ __forceinline__ uint16_t f2bf(float x) {
  // round-to-nearest-even float32 -> bf16 bits
  uint32_t u = __float_as_uint(x);
  u += 0x7FFFu + ((u >> 16) & 1u);
  return (uint16_t)(u >> 16);
}

union FragU { v16bf v; uint4 u[2]; };

// Load a 16x32 bf16 fragment (A layout; B uses the mirrored layout with the
// lane indexing the output column, i.e. a row of the k-contiguous operand).
// Lane L: row = L&15, khalf = L>>4. VGPR0..3 hold K = khalf*8..+7,
// VGPR4..7 hold K = 16+khalf*8..+7  -> two contiguous 16B chunks per lane.
__device__ __forceinline__ v16bf load_frag(const uint16_t* __restrict__ base,
                                           int ldk, int lane) {
  int r  = lane & 15;
  int kh = lane >> 4;
  const uint16_t* p = base + (size_t)r * ldk + kh * 8;
  FragU f;
  f.u[0] = *(const uint4*)(p);        // K = kh*8 .. kh*8+7
  f.u[1] = *(const uint4*)(p + 16);   // K = 16+kh*8 .. +7
  return f.v;
}

__device__ __forceinline__ v8f wmma_bf16(v16bf a, v16bf b, v8f c) {
  // D(16x16,f32) = A(16x32,bf16) * B(32x16,bf16) + C
  return __builtin_amdgcn_wmma_f32_16x16x32_bf16(
      /*neg_a=*/false, a, /*neg_b=*/false, b,
      /*c_mod=*/(short)0, c, /*reuse_a=*/false, /*reuse_b=*/false);
}

// ---------------------------------------------------------------- kernels ---

// fp32 -> bf16 weight conversion (once per launch; weights then L2-resident)
__global__ __launch_bounds__(256) void cvt_kernel(const float* __restrict__ src,
                                                  uint16_t* __restrict__ dst,
                                                  int n) {
  int i = blockIdx.x * 256 + threadIdx.x;
  if (i < n) dst[i] = f2bf(src[i]);
}

// GroupNorm over (C/G, H, W) per (b, g); writes bf16 tokens h[b][hw][c].
__global__ __launch_bounds__(256) void gn_kernel(const float* __restrict__ x,
                                                 const float* __restrict__ gamma,
                                                 const float* __restrict__ beta,
                                                 uint16_t* __restrict__ h) {
  __shared__ float s_sum[256];
  __shared__ float s_sq[256];
  int b = blockIdx.x / G_;
  int g = blockIdx.x % G_;
  const float* xb = x + (size_t)(b * C_ + g * CG_) * HW_;
  const int elems = CG_ * HW_;                       // 65536
  float sum = 0.f, sq = 0.f;
  for (int i = threadIdx.x * 4; i < elems; i += 256 * 4) {
    float4 v = *(const float4*)(xb + i);
    sum += v.x + v.y + v.z + v.w;
    sq  += v.x * v.x + v.y * v.y + v.z * v.z + v.w * v.w;
  }
  s_sum[threadIdx.x] = sum;
  s_sq[threadIdx.x]  = sq;
  __syncthreads();
  for (int off = 128; off > 0; off >>= 1) {
    if (threadIdx.x < off) {
      s_sum[threadIdx.x] += s_sum[threadIdx.x + off];
      s_sq[threadIdx.x]  += s_sq[threadIdx.x + off];
    }
    __syncthreads();
  }
  float mean = s_sum[0] * (1.0f / elems);
  float var  = s_sq[0] * (1.0f / elems) - mean * mean;
  float rstd = rsqrtf(var + EPS_);
  for (int i = threadIdx.x * 4; i < elems; i += 256 * 4) {
    int cl = i >> 10;                 // i / HW_
    int hw = i & (HW_ - 1);
    int c  = g * CG_ + cl;
    float ga = gamma[c] * rstd;
    float be = beta[c] - mean * ga;   // y = x*ga + be
    float4 v = *(const float4*)(xb + i);
    uint16_t* hp = h + ((size_t)b * HW_ + hw) * C_ + c;   // token-major layout
    hp[0 * C_] = f2bf(v.x * ga + be);
    hp[1 * C_] = f2bf(v.y * ga + be);
    hp[2 * C_] = f2bf(v.z * ga + be);
    hp[3 * C_] = f2bf(v.w * ga + be);
  }
}

// q/k/v = h @ W{q,k,v}^T + b. Each wave: 16 tokens x 64 channels for all of
// q,k,v (12 accumulators); one A fragment feeds 12 WMMAs per K-step. The 8
// waves of a block share the same weight quad (dq) -> B tiles hit WGP$/L2,
// and cross-wave occupancy hides the per-wave load->wmma dependency.
// q,k stored [b][n][d]; v stored transposed [b][d][n] for the PV GEMM.
__global__ __launch_bounds__(256) void qkv_kernel(
    const uint16_t* __restrict__ h,
    const uint16_t* __restrict__ Wq16, const uint16_t* __restrict__ Wk16,
    const uint16_t* __restrict__ Wv16,
    const float* __restrict__ bq, const float* __restrict__ bk,
    const float* __restrict__ bv,
    uint16_t* __restrict__ q, uint16_t* __restrict__ k,
    uint16_t* __restrict__ vT) {
  int lane = threadIdx.x & 31;
  int wave = threadIdx.x >> 5;
  // grid = B * (C/64) * (N/(16*8)) = 16*8*8 = 1024 blocks
  int blk = blockIdx.x;
  int b   = blk >> 6;
  int dq  = (blk >> 3) & 7;                         // channel quad: 4 d-tiles
  int nt  = ((blk & 7) << 3) + wave;                // token tile 0..63
  const uint16_t* ha  = h    + ((size_t)b * HW_ + nt * 16) * C_;
  const uint16_t* wqp = Wq16 + (size_t)dq * 64 * C_;
  const uint16_t* wkp = Wk16 + (size_t)dq * 64 * C_;
  const uint16_t* wvp = Wv16 + (size_t)dq * 64 * C_;
  v8f aq[4] = {}, ak[4] = {}, av[4] = {};
  for (int kk = 0; kk < C_; kk += 32) {
    if (kk + 64 < C_) __builtin_prefetch(wqp + kk + 64, 0, 0);  // global_prefetch
    v16bf A = load_frag(ha + kk, C_, lane);
#pragma unroll
    for (int j = 0; j < 4; ++j) {
      aq[j] = wmma_bf16(A, load_frag(wqp + (size_t)j * 16 * C_ + kk, C_, lane), aq[j]);
      ak[j] = wmma_bf16(A, load_frag(wkp + (size_t)j * 16 * C_ + kk, C_, lane), ak[j]);
      av[j] = wmma_bf16(A, load_frag(wvp + (size_t)j * 16 * C_ + kk, C_, lane), av[j]);
    }
  }
  int n  = lane & 15;
  int kh = lane >> 4;
#pragma unroll
  for (int j = 0; j < 4; ++j) {
    int d = dq * 64 + j * 16 + n;
    float biq = bq[d], bik = bk[d], biv = bv[d];
#pragma unroll
    for (int rr = 0; rr < 8; ++rr) {
      int m = rr + kh * 8;
      size_t row = (size_t)b * HW_ + nt * 16 + m;
      q[row * C_ + d] = f2bf(aq[j][rr] + biq);
      k[row * C_ + d] = f2bf(ak[j][rr] + bik);
      vT[((size_t)b * C_ + d) * HW_ + nt * 16 + m] = f2bf(av[j][rr] + biv);
    }
  }
}

// S[b][n][m] = scale * q[b][n][:] . k[b][m][:]
// Wave: 16 queries x 64 keys (4 accumulators, A fragment reused 4x).
__global__ __launch_bounds__(256) void qk_kernel(const uint16_t* __restrict__ q,
                                                 const uint16_t* __restrict__ k,
                                                 float* __restrict__ S) {
  int lane = threadIdx.x & 31;
  int wave = threadIdx.x >> 5;
  // grid = B * (N/64) * (N/(16*8)) = 16*16*8 = 2048 blocks
  int blk = blockIdx.x;
  int b   = blk >> 7;
  int mq  = (blk >> 3) & 15;                        // key quad: 4 m-tiles
  int nt  = ((blk & 7) << 3) + wave;                // query tile 0..63
  const uint16_t* qa = q + ((size_t)b * HW_ + nt * 16) * C_;
  const uint16_t* kb = k + ((size_t)b * HW_ + mq * 64) * C_;
  v8f acc[4] = {};
  for (int kk = 0; kk < C_; kk += 32) {
    v16bf A = load_frag(qa + kk, C_, lane);
#pragma unroll
    for (int j = 0; j < 4; ++j)
      acc[j] = wmma_bf16(A, load_frag(kb + (size_t)j * 16 * C_ + kk, C_, lane), acc[j]);
  }
  const float scale = 0.0441941738241592f;          // 512^-0.5
  int n  = lane & 15;
  int kh = lane >> 4;
#pragma unroll
  for (int j = 0; j < 4; ++j) {
#pragma unroll
    for (int rr = 0; rr < 8; ++rr) {
      int m = rr + kh * 8;                          // query row within tile
      S[((size_t)b * HW_ + nt * 16 + m) * HW_ + mq * 64 + j * 16 + n] =
          acc[j][rr] * scale;
    }
  }
}

// Row softmax over 1024 keys; writes bf16 probabilities P.
__global__ __launch_bounds__(256) void softmax_kernel(const float* __restrict__ S,
                                                      uint16_t* __restrict__ P) {
  __shared__ float red[256];
  size_t row = blockIdx.x;                 // B*N rows
  const float* s = S + row * HW_;
  float4 v = *(const float4*)(s + threadIdx.x * 4);
  float mx = fmaxf(fmaxf(v.x, v.y), fmaxf(v.z, v.w));
  red[threadIdx.x] = mx;
  __syncthreads();
  for (int off = 128; off > 0; off >>= 1) {
    if (threadIdx.x < off)
      red[threadIdx.x] = fmaxf(red[threadIdx.x], red[threadIdx.x + off]);
    __syncthreads();
  }
  mx = red[0];
  __syncthreads();
  float e0 = __expf(v.x - mx), e1 = __expf(v.y - mx);
  float e2 = __expf(v.z - mx), e3 = __expf(v.w - mx);
  red[threadIdx.x] = e0 + e1 + e2 + e3;
  __syncthreads();
  for (int off = 128; off > 0; off >>= 1) {
    if (threadIdx.x < off) red[threadIdx.x] += red[threadIdx.x + off];
    __syncthreads();
  }
  float inv = 1.0f / red[0];
  uint16_t* p = P + row * HW_ + threadIdx.x * 4;
  p[0] = f2bf(e0 * inv);
  p[1] = f2bf(e1 * inv);
  p[2] = f2bf(e2 * inv);
  p[3] = f2bf(e3 * inv);
}

// O[b][n][d] = sum_m P[b][n][m] * V[b][m][d]   (V stored transposed: vT[b][d][m])
// Wave: 16 tokens x 64 channels (4 accumulators).
__global__ __launch_bounds__(256) void pv_kernel(const uint16_t* __restrict__ P,
                                                 const uint16_t* __restrict__ vT,
                                                 uint16_t* __restrict__ O) {
  int lane = threadIdx.x & 31;
  int wave = threadIdx.x >> 5;
  // grid = B * (C/64) * (N/(16*8)) = 16*8*8 = 1024 blocks
  int blk = blockIdx.x;
  int b   = blk >> 6;
  int dq  = (blk >> 3) & 7;
  int nt  = ((blk & 7) << 3) + wave;
  const uint16_t* pa = P  + ((size_t)b * HW_ + nt * 16) * HW_;
  const uint16_t* vb = vT + ((size_t)b * C_  + dq * 64) * HW_;
  v8f acc[4] = {};
  for (int kk = 0; kk < HW_; kk += 32) {
    v16bf A = load_frag(pa + kk, HW_, lane);
#pragma unroll
    for (int j = 0; j < 4; ++j)
      acc[j] = wmma_bf16(A, load_frag(vb + (size_t)j * 16 * HW_ + kk, HW_, lane), acc[j]);
  }
  int n  = lane & 15;
  int kh = lane >> 4;
#pragma unroll
  for (int j = 0; j < 4; ++j) {
#pragma unroll
    for (int rr = 0; rr < 8; ++rr) {
      int m = rr + kh * 8;
      O[((size_t)b * HW_ + nt * 16 + m) * C_ + dq * 64 + j * 16 + n] =
          f2bf(acc[j][rr]);
    }
  }
}

// Y = O @ Wp^T + bp ; out[b][c][hw] = x[b][c][hw] + Y[b][hw][c]  (fused
// transpose + residual; lane writes 8 hw-consecutive floats for fixed c).
// Wave: 16 tokens x 64 channels (4 accumulators).
__global__ __launch_bounds__(256) void proj_kernel(const uint16_t* __restrict__ O,
                                                   const uint16_t* __restrict__ Wp16,
                                                   const float* __restrict__ bp,
                                                   const float* __restrict__ x,
                                                   float* __restrict__ out) {
  int lane = threadIdx.x & 31;
  int wave = threadIdx.x >> 5;
  // grid = B * (C/64) * (N/(16*8)) = 16*8*8 = 1024 blocks
  int blk = blockIdx.x;
  int b   = blk >> 6;
  int dq  = (blk >> 3) & 7;
  int nt  = ((blk & 7) << 3) + wave;
  const uint16_t* oa = O    + ((size_t)b * HW_ + nt * 16) * C_;
  const uint16_t* wb = Wp16 + (size_t)dq * 64 * C_;
  v8f acc[4] = {};
  for (int kk = 0; kk < C_; kk += 32) {
    if (kk + 64 < C_) __builtin_prefetch(wb + kk + 64, 0, 0);
    v16bf A = load_frag(oa + kk, C_, lane);
#pragma unroll
    for (int j = 0; j < 4; ++j)
      acc[j] = wmma_bf16(A, load_frag(wb + (size_t)j * 16 * C_ + kk, C_, lane), acc[j]);
  }
  int kh = lane >> 4;
#pragma unroll
  for (int j = 0; j < 4; ++j) {
    int e = dq * 64 + j * 16 + (lane & 15);         // output channel
    float bias = bp[e];
#pragma unroll
    for (int rr = 0; rr < 8; ++rr) {
      int hw = nt * 16 + kh * 8 + rr;               // consecutive along rr
      size_t idx = ((size_t)b * C_ + e) * HW_ + hw;
      out[idx] = x[idx] + acc[j][rr] + bias;
    }
  }
}

// ----------------------------------------------------------------- launch ---

extern "C" void kernel_launch(void* const* d_in, const int* in_sizes, int n_in,
                              void* d_out, int out_size, void* d_ws, size_t ws_size,
                              hipStream_t stream) {
  const float* x     = (const float*)d_in[0];
  const float* gam   = (const float*)d_in[1];
  const float* bet   = (const float*)d_in[2];
  const float* Wq    = (const float*)d_in[3];
  const float* bq    = (const float*)d_in[4];
  const float* Wk    = (const float*)d_in[5];
  const float* bk    = (const float*)d_in[6];
  const float* Wv    = (const float*)d_in[7];
  const float* bv    = (const float*)d_in[8];
  const float* Wp    = (const float*)d_in[9];
  const float* bp    = (const float*)d_in[10];
  float*       out   = (float*)d_out;

  const size_t MB = 1ull << 20;
  char* ws = (char*)d_ws;
  uint16_t* h    = (uint16_t*)(ws + 0 * MB);      // 16 MB  bf16 [B][N][C]
  uint16_t* qb   = (uint16_t*)(ws + 16 * MB);     // 16 MB
  uint16_t* kb   = (uint16_t*)(ws + 32 * MB);     // 16 MB
  uint16_t* vT   = (uint16_t*)(ws + 48 * MB);     // 16 MB  bf16 [B][C][N]
  float*    S    = (float*)   (ws + 64 * MB);     // 64 MB  f32  [B][N][N]
  uint16_t* P    = (uint16_t*)(ws + 128 * MB);    // 32 MB  bf16 [B][N][N]
  uint16_t* O    = (uint16_t*)(ws + 160 * MB);    // 16 MB
  uint16_t* Wq16 = (uint16_t*)(ws + 176 * MB);    // 0.5 MB each
  uint16_t* Wk16 = Wq16 + (size_t)C_ * C_;
  uint16_t* Wv16 = Wk16 + (size_t)C_ * C_;
  uint16_t* Wp16 = Wv16 + (size_t)C_ * C_;

  const int wn = C_ * C_;                         // 262144
  cvt_kernel<<<(wn + 255) / 256, 256, 0, stream>>>(Wq, Wq16, wn);
  cvt_kernel<<<(wn + 255) / 256, 256, 0, stream>>>(Wk, Wk16, wn);
  cvt_kernel<<<(wn + 255) / 256, 256, 0, stream>>>(Wv, Wv16, wn);
  cvt_kernel<<<(wn + 255) / 256, 256, 0, stream>>>(Wp, Wp16, wn);

  gn_kernel<<<B_ * G_, 256, 0, stream>>>(x, gam, bet, h);

  qkv_kernel<<<1024, 256, 0, stream>>>(h, Wq16, Wk16, Wv16, bq, bk, bv,
                                       qb, kb, vT);
  qk_kernel<<<2048, 256, 0, stream>>>(qb, kb, S);

  softmax_kernel<<<B_ * HW_, 256, 0, stream>>>(S, P);

  pv_kernel<<<1024, 256, 0, stream>>>(P, vT, O);

  proj_kernel<<<1024, 256, 0, stream>>>(O, Wp16, bp, x, out);
}
